// CSwinAttention_block_60352880443970
// MI455X (gfx1250) — compile-verified, matrix operand order checked by reading, not run on hardware
//
#include <hip/hip_runtime.h>

typedef __attribute__((ext_vector_type(16))) __bf16       v16bf;
typedef __attribute__((ext_vector_type(8)))  float        v8f;
typedef __attribute__((ext_vector_type(8)))  unsigned int v8u;
typedef __attribute__((ext_vector_type(4)))  unsigned int v4u;

#define BATCH   8
#define CDIMF   256
#define HH      64
#define WW      64
#define LTOK    4096
#define MTOK    32768
#define NHEADS  4
#define CHALF   128
#define CHEAD   32
#define WINSZ   256
#define NWIN    128

// ---- CDNA5 async LDS<->global helpers (ASYNCcnt-tracked, ISA 08_async_tensor) ----
template <int N>
__device__ inline void wait_async() {
#if __has_builtin(__builtin_amdgcn_s_wait_asynccnt)
  __builtin_amdgcn_s_wait_asynccnt((short)N);
#else
  asm volatile("s_wait_asynccnt %0" ::"i"(N) : "memory");
#endif
}
// GVS mode: lds_byte_off (VGPR), global byte offset (VGPR i32), base (SGPR pair)
__device__ inline void async_ld_b128(unsigned ldsoff, unsigned goff, const void* sbase) {
  asm volatile("global_load_async_to_lds_b128 %0, %1, %2"
               :: "v"(ldsoff), "v"(goff), "s"(sbase) : "memory");
}
__device__ inline void async_st_b128(unsigned goff, unsigned ldsoff, const void* sbase) {
  asm volatile("global_store_async_from_lds_b128 %0, %1, %2"
               :: "v"(goff), "v"(ldsoff), "s"(sbase) : "memory");
}
// Opaque zero: compiler cannot fold it, used to keep async-written LDS "visibly stored"
// (otherwise LLVM proves the shared array has no stores and folds ds_loads to undef).
__device__ inline unsigned opaque_zero() {
  unsigned z;
  asm volatile("s_mov_b32 %0, 0" : "=s"(z));
  return z;
}

// Load a 16x32 bf16 WMMA A/B fragment from a 32-element contiguous K-row in LDS.
// Lanes 0-15 hold K {0..7,16..23}, lanes 16-31 hold K {8..15,24..31} (ISA 7.12.2).
__device__ inline v16bf frag_ld_bf16(const __bf16* rowp, int lane) {
  const __bf16* p = rowp + ((lane & 16) ? 8 : 0);
  v4u lo = *(const v4u*)(p);
  v4u hi = *(const v4u*)(p + 16);
  v8u u;
  u[0] = lo[0]; u[1] = lo[1]; u[2] = lo[2]; u[3] = lo[3];
  u[4] = hi[0]; u[5] = hi[1]; u[6] = hi[2]; u[7] = hi[3];
  return __builtin_bit_cast(v16bf, u);
}

__device__ inline v8f wmma_bf16(v16bf a, v16bf b, v8f c) {
  return __builtin_amdgcn_wmma_f32_16x16x32_bf16(false, a, false, b, (short)0, c, false, false);
}

// ---------------- LayerNorm + bf16 cast (token-major [M, C]) ----------------
__global__ __launch_bounds__(256) void ln_cast_kernel(
    const float* __restrict__ x, const float* __restrict__ g,
    const float* __restrict__ b, __bf16* __restrict__ xn) {
  int m = blockIdx.x;
  int c = threadIdx.x;
  int bb = m >> 12;
  int yx = m & 4095;
  float v = x[(bb * CDIMF + c) * LTOK + yx];
  float s = v, s2 = v * v;
#pragma unroll
  for (int o = 16; o > 0; o >>= 1) {
    s  += __shfl_down(s,  o, 32);
    s2 += __shfl_down(s2, o, 32);
  }
  __shared__ float ps[8], ps2[8], stat[2];
  int wid = c >> 5, lane = c & 31;
  if (lane == 0) { ps[wid] = s; ps2[wid] = s2; }
  __syncthreads();
  if (c == 0) {
    float ts = 0.f, ts2 = 0.f;
    for (int i = 0; i < 8; ++i) { ts += ps[i]; ts2 += ps2[i]; }
    float mu  = ts  * (1.0f / CDIMF);
    float var = ts2 * (1.0f / CDIMF) - mu * mu;
    stat[0] = mu;
    stat[1] = rsqrtf(var + 1e-4f);
  }
  __syncthreads();
  float r = (v - stat[0]) * stat[1] * g[c] + b[c];
  xn[m * CDIMF + c] = (__bf16)r;
}

// ---------------- f32 -> bf16 cast ----------------
__global__ void cvt_f32_bf16(const float* __restrict__ in, __bf16* __restrict__ out, int n) {
  int i = blockIdx.x * blockDim.x + threadIdx.x;
  if (i < n) out[i] = (__bf16)in[i];
}

// ---------------- QKV GEMM: [M,256] x [768,256]^T -> bf16 [M,768] ----------------
// Double-buffered, async LDS staging.
__global__ __launch_bounds__(256) void gemm_qkv_kernel(
    const __bf16* __restrict__ A, const __bf16* __restrict__ Wt,
    __bf16* __restrict__ Out) {
  __shared__ __bf16 lds[2][4096];  // per buffer: [0..2047]=A 64x32, [2048..4095]=W 64x32
  {  // keep async-written LDS visibly stored (never executes, unprovable)
    unsigned oz = opaque_zero();
    if (oz) ((__bf16*)lds)[oz] = (__bf16)1.0f;
  }
  const int Kdim = 256, Ndim = 768;
  int m0 = blockIdx.y * 64, n0 = blockIdx.x * 64;
  int t = threadIdx.x, lane = t & 31, wave = t >> 5;
  int srow = wave >> 1;
  int scol = (wave & 1) * 2;
  int lrow = t >> 2, lq = (t & 3) * 8;
  const char* lbase = (const char*)&lds[0][0];
  unsigned la = (unsigned)((const char*)&lds[0][lrow * 32 + lq] - lbase);
  unsigned lw = la + 2048u * 2u;
  unsigned aoff = (unsigned)(((m0 + lrow) * Kdim + lq) * 2);
  unsigned woff = (unsigned)(((n0 + lrow) * Kdim + lq) * 2);

  async_ld_b128(la, aoff, A);
  async_ld_b128(lw, woff, Wt);
  v8f acc0 = {}; v8f acc1 = {};
  int buf = 0;
  for (int kk = 0; kk < Kdim; kk += 32, buf ^= 1) {
    if (kk + 32 < Kdim) {
      unsigned nb = (unsigned)(buf ^ 1) * 8192u;
      async_ld_b128(la + nb, aoff + (unsigned)(kk + 32) * 2u, A);
      async_ld_b128(lw + nb, woff + (unsigned)(kk + 32) * 2u, Wt);
      wait_async<2>();   // previous buffer's 2 loads done; the 2 just issued may fly
    } else {
      wait_async<0>();
    }
    __syncthreads();
    const __bf16* Ab = &lds[buf][0];
    const __bf16* Wb = &lds[buf][2048];
    v16bf af = frag_ld_bf16(&Ab[(srow * 16 + (lane & 15)) * 32], lane);
    v16bf b0 = frag_ld_bf16(&Wb[( scol      * 16 + (lane & 15)) * 32], lane);
    v16bf b1 = frag_ld_bf16(&Wb[((scol + 1) * 16 + (lane & 15)) * 32], lane);
    acc0 = wmma_bf16(af, b0, acc0);
    acc1 = wmma_bf16(af, b1, acc1);
    __syncthreads();
  }
  int rowoff = (lane & 16) ? 8 : 0;
  int ncol0 = n0 + scol * 16 + (lane & 15);
#pragma unroll
  for (int r = 0; r < 8; ++r) {
    size_t mrow = (size_t)(m0 + srow * 16 + rowoff + r);
    Out[mrow * Ndim + ncol0]      = (__bf16)acc0[r];
    Out[mrow * Ndim + ncol0 + 16] = (__bf16)acc1[r];
  }
}

// ---------------- Stripe-window attention, softmax, LePE ----------------
__global__ __launch_bounds__(256) void attn_kernel(
    const __bf16* __restrict__ qkv,
    const float* __restrict__ lw1, const float* __restrict__ lb1,
    const float* __restrict__ lw2, const float* __restrict__ lb2,
    float* __restrict__ attn_out,
    __bf16* __restrict__ xcat) {
  extern __shared__ char smem[];
  float*  S    = (float*)smem;                  // 256x256 f32 scores (256 KB)
  __bf16* qs   = (__bf16*)(smem + 65536 * 4);   // [pos][ci]
  __bf16* ks   = qs + WINSZ * CHEAD;            // [pos][ci]
  __bf16* vs_t = ks + WINSZ * CHEAD;            // [ci][pos]  (transposed for B-fragments)

  int hd = blockIdx.x, win = blockIdx.y, br = blockIdx.z;
  int bb = win >> 4, blk = win & 15;
  int hsp = br ? 4 : 64;
  int wsp = br ? 64 : 4;
  int t = threadIdx.x, lane = t & 31, wave = t >> 5;

  // ---- load q (pre-scaled), k, v-transposed for this (window, head) ----
  {
    int p = t;
    int hin = p / wsp, wi = p % wsp;
    int y = br ? (blk * 4 + hin) : hin;
    int x = br ? wi : (blk * 4 + wi);
    int tok = y * WW + x;
    const __bf16* base = qkv + (size_t)(bb * LTOK + tok) * 768 + br * CHALF + hd;
#pragma unroll
    for (int ci = 0; ci < CHEAD; ++ci) {
      int ch = ci * 4;  // channel layout (c, head): stride NHEADS
      qs[p * CHEAD + ci]   = (__bf16)((float)base[ch] * 0.125f);  // scale = (C/heads)^-0.5
      ks[p * CHEAD + ci]   = base[256 + ch];
      vs_t[ci * WINSZ + p] = base[512 + ch];
    }
  }
  __syncthreads();

  // ---- S = (q*scale) @ k^T : wave owns rows [32*wave, 32*wave+32) ----
  {
    int rt0 = wave * 2;
#pragma unroll
    for (int rr = 0; rr < 2; ++rr) {
      v16bf af = frag_ld_bf16(&qs[((rt0 + rr) * 16 + (lane & 15)) * CHEAD], lane);
      for (int ct = 0; ct < 16; ++ct) {
        v16bf bf = frag_ld_bf16(&ks[(ct * 16 + (lane & 15)) * CHEAD], lane);
        v8f d = {};
        d = wmma_bf16(af, bf, d);
        int rowoff = (lane & 16) ? 8 : 0;
        int colg = ct * 16 + (lane & 15);
#pragma unroll
        for (int r = 0; r < 8; ++r)
          S[((rt0 + rr) * 16 + rowoff + r) * WINSZ + colg] = d[r];
      }
    }
  }
  __syncthreads();

  // ---- softmax: one row per thread, in LDS ----
  {
    float* row = S + t * WINSZ;
    float mx = -3.4e38f;
    for (int j = 0; j < WINSZ; ++j) mx = fmaxf(mx, row[j]);
    float sum = 0.f;
    for (int j = 0; j < WINSZ; ++j) { float e = __expf(row[j] - mx); row[j] = e; sum += e; }
    float inv = 1.0f / sum;
    for (int j = 0; j < WINSZ; ++j) row[j] *= inv;
  }
  __syncthreads();

  // ---- attention matrix -> HBM via async LDS->global (no VGPR round-trip) ----
  {
    float* dst = attn_out + (size_t)br * ((size_t)NWIN * NHEADS * WINSZ * WINSZ)
               + (size_t)(win * NHEADS + hd) * (WINSZ * WINSZ);
    unsigned sbase = (unsigned)((char*)S - smem);  // 0
    for (int c = 0; c < 64; ++c) {
      unsigned off = (unsigned)(t + c * 256) * 16u;
      async_st_b128(off, sbase + off, dst);
    }
    wait_async<0>();  // stores must finish before P is packed over the f32 data
  }
  __syncthreads();

  // ---- pack P rows to bf16 in place (row start): A-fragments become ds_load_b128 ----
  {
    float* row = S + t * WINSZ;
    __bf16* prow = (__bf16*)row;
    for (int j0 = 0; j0 < WINSZ; j0 += 8) {
      float tmp[8];
#pragma unroll
      for (int jj = 0; jj < 8; ++jj) tmp[jj] = row[j0 + jj];
      asm volatile("" ::: "memory");  // keep reads ahead of aliased bf16 writes
#pragma unroll
      for (int jj = 0; jj < 8; ++jj) prow[j0 + jj] = (__bf16)tmp[jj];
    }
  }
  __syncthreads();

  // ---- O = P @ V entirely from LDS (packed-bf16 P rows, transposed V) ----
  v8f acc[2][2];
#pragma unroll
  for (int a = 0; a < 2; ++a) { acc[a][0] = (v8f){}; acc[a][1] = (v8f){}; }
  {
    int rt0 = wave * 2;
    for (int kt = 0; kt < 8; ++kt) {
      v16bf bfr0 = frag_ld_bf16(&vs_t[( (lane & 15))       * WINSZ + kt * 32], lane);
      v16bf bfr1 = frag_ld_bf16(&vs_t[((lane & 15) + 16)   * WINSZ + kt * 32], lane);
#pragma unroll
      for (int rr = 0; rr < 2; ++rr) {
        const __bf16* prow = (const __bf16*)(S + ((rt0 + rr) * 16 + (lane & 15)) * WINSZ);
        v16bf af = frag_ld_bf16(prow + kt * 32, lane);
        acc[rr][0] = wmma_bf16(af, bfr0, acc[rr][0]);
        acc[rr][1] = wmma_bf16(af, bfr1, acc[rr][1]);
      }
    }
  }
  __syncthreads();  // all P reads complete before S is reused for O

  // ---- stash O compactly into LDS ----
  {
    int rt0 = wave * 2;
    int rowoff = (lane & 16) ? 8 : 0;
#pragma unroll
    for (int rr = 0; rr < 2; ++rr)
#pragma unroll
      for (int nt = 0; nt < 2; ++nt) {
        int ci = nt * 16 + (lane & 15);
#pragma unroll
        for (int r = 0; r < 8; ++r)
          S[((rt0 + rr) * 16 + rowoff + r) * CHEAD + ci] = acc[rr][nt][r];
      }
  }
  __syncthreads();

  // ---- LePE (depthwise 3x3 inside window) + write xcat token-major bf16 ----
  {
    int p = t;
    int hin = p / wsp, wi = p % wsp;
    int y = br ? (blk * 4 + hin) : hin;
    int x = br ? wi : (blk * 4 + wi);
    int tok = y * WW + x;
    const float* lw = br ? lw2 : lw1;
    const float* lb = br ? lb2 : lb1;
    __bf16* dst = xcat + (size_t)(bb * LTOK + tok) * CDIMF + br * CHALF + hd;
    for (int ci = 0; ci < CHEAD; ++ci) {
      int ch = ci * 4 + hd;
      float a = lb[ch];
#pragma unroll
      for (int dy = -1; dy <= 1; ++dy)
#pragma unroll
        for (int dx = -1; dx <= 1; ++dx) {
          int nh = hin + dy, nw = wi + dx;
          if (nh >= 0 && nh < hsp && nw >= 0 && nw < wsp)
            a += lw[ch * 9 + (dy + 1) * 3 + (dx + 1)] * (float)vs_t[ci * WINSZ + nh * wsp + nw];
        }
      dst[ci * 4] = (__bf16)(S[p * CHEAD + ci] + a);
    }
  }
}

// ---------------- Projection GEMM: [M,256] x [256,256]^T + bias -> f32 [B,C,H,W] ----------------
__global__ __launch_bounds__(256) void gemm_proj_kernel(
    const __bf16* __restrict__ A, const __bf16* __restrict__ Wt,
    const float* __restrict__ bias, float* __restrict__ Out) {
  __shared__ __bf16 lds[2][4096];
  {  // keep async-written LDS visibly stored (never executes, unprovable)
    unsigned oz = opaque_zero();
    if (oz) ((__bf16*)lds)[oz] = (__bf16)1.0f;
  }
  const int Kdim = 256;
  int m0 = blockIdx.y * 64, n0 = blockIdx.x * 64;
  int t = threadIdx.x, lane = t & 31, wave = t >> 5;
  int srow = wave >> 1;
  int scol = (wave & 1) * 2;
  int lrow = t >> 2, lq = (t & 3) * 8;
  const char* lbase = (const char*)&lds[0][0];
  unsigned la = (unsigned)((const char*)&lds[0][lrow * 32 + lq] - lbase);
  unsigned lw = la + 2048u * 2u;
  unsigned aoff = (unsigned)(((m0 + lrow) * Kdim + lq) * 2);
  unsigned woff = (unsigned)(((n0 + lrow) * Kdim + lq) * 2);

  async_ld_b128(la, aoff, A);
  async_ld_b128(lw, woff, Wt);
  v8f acc0 = {}; v8f acc1 = {};
  int buf = 0;
  for (int kk = 0; kk < Kdim; kk += 32, buf ^= 1) {
    if (kk + 32 < Kdim) {
      unsigned nb = (unsigned)(buf ^ 1) * 8192u;
      async_ld_b128(la + nb, aoff + (unsigned)(kk + 32) * 2u, A);
      async_ld_b128(lw + nb, woff + (unsigned)(kk + 32) * 2u, Wt);
      wait_async<2>();
    } else {
      wait_async<0>();
    }
    __syncthreads();
    const __bf16* Ab = &lds[buf][0];
    const __bf16* Wb = &lds[buf][2048];
    v16bf af = frag_ld_bf16(&Ab[(srow * 16 + (lane & 15)) * 32], lane);
    v16bf b0 = frag_ld_bf16(&Wb[( scol      * 16 + (lane & 15)) * 32], lane);
    v16bf b1 = frag_ld_bf16(&Wb[((scol + 1) * 16 + (lane & 15)) * 32], lane);
    acc0 = wmma_bf16(af, b0, acc0);
    acc1 = wmma_bf16(af, b1, acc1);
    __syncthreads();
  }
  int rowoff = (lane & 16) ? 8 : 0;
  int n0c = n0 + scol * 16 + (lane & 15);
  float b0v = bias[n0c], b1v = bias[n0c + 16];
#pragma unroll
  for (int r = 0; r < 8; ++r) {
    int mrow = m0 + srow * 16 + rowoff + r;
    int bb = mrow >> 12, yx = mrow & 4095;
    Out[((size_t)bb * CDIMF + n0c)      * LTOK + yx] = acc0[r] + b0v;
    Out[((size_t)bb * CDIMF + n0c + 16) * LTOK + yx] = acc1[r] + b1v;
  }
}

extern "C" void kernel_launch(void* const* d_in, const int* in_sizes, int n_in,
                              void* d_out, int out_size, void* d_ws, size_t ws_size,
                              hipStream_t stream) {
  const float* x      = (const float*)d_in[0];
  const float* ln_g   = (const float*)d_in[1];
  const float* ln_b   = (const float*)d_in[2];
  const float* w_qkv  = (const float*)d_in[3];
  const float* proj_w = (const float*)d_in[4];
  const float* proj_b = (const float*)d_in[5];
  const float* lw1    = (const float*)d_in[6];
  const float* lb1    = (const float*)d_in[7];
  const float* lw2    = (const float*)d_in[8];
  const float* lb2    = (const float*)d_in[9];
  float* out = (float*)d_out;

  char* ws = (char*)d_ws;
  __bf16* xn     = (__bf16*)(ws);                                            // 16 MB
  __bf16* wqkvb  = (__bf16*)(ws + 16777216);                                 // 384 KB
  __bf16* projwb = (__bf16*)(ws + 16777216 + 393216);                        // 128 KB
  __bf16* qkvb   = (__bf16*)(ws + 16777216 + 393216 + 131072);               // 48 MB
  __bf16* xcat   = (__bf16*)(ws + 16777216 + 393216 + 131072 + 50331648);    // 16 MB

  ln_cast_kernel<<<MTOK, 256, 0, stream>>>(x, ln_g, ln_b, xn);
  cvt_f32_bf16<<<(768 * 256) / 256, 256, 0, stream>>>(w_qkv, wqkvb, 768 * 256);
  cvt_f32_bf16<<<(256 * 256) / 256, 256, 0, stream>>>(proj_w, projwb, 256 * 256);
  gemm_qkv_kernel<<<dim3(12, 512), 256, 0, stream>>>(xn, wqkvb, qkvb);

  size_t smem = (size_t)65536 * 4 + (size_t)3 * WINSZ * CHEAD * 2;  // 311,296 B < 320 KB WGP LDS
  (void)hipFuncSetAttribute(reinterpret_cast<const void*>(attn_kernel),
                            hipFuncAttributeMaxDynamicSharedMemorySize, (int)smem);
  attn_kernel<<<dim3(NHEADS, NWIN, 2), 256, smem, stream>>>(
      qkvb, lw1, lb1, lw2, lb2, out + 8388608, xcat);

  gemm_proj_kernel<<<dim3(4, 512), 256, 0, stream>>>(xcat, projwb, proj_b, out);
}
